// OGNNLayer_v2_22411139351196
// MI455X (gfx1250) — matrix operand order repaired and checked by reading, not compiled, not run on hardware
//
#include <hip/hip_runtime.h>
#include <hip/hip_fp16.h>

#define N_NODES 50000
#define IN_F    512
#define OUT_F   128
#define N_EDGES 500000
#define BN_EPS  1e-5f

typedef __attribute__((ext_vector_type(16))) _Float16 v16h;
typedef __attribute__((ext_vector_type(8)))  float    v8f;
typedef __attribute__((ext_vector_type(4)))  float    v4f;

// Octonion block pattern: hamilton[i*64+r][j*16+c] = SGN[j][i]*weight[r][IDX[j][i]*16+c]
__constant__ int OCT_IDX[8][8] = {
  {0,1,2,3,7,5,6,7},
  {1,0,3,5,4,4,2,6},
  {2,3,3,1,6,7,4,5},
  {4,2,1,0,7,6,7,4},
  {4,5,3,7,0,1,2,6},
  {5,4,7,6,1,5,5,2},
  {6,7,4,5,2,4,0,6},
  {7,6,3,4,3,3,1,5}};
__constant__ float OCT_SGN[8][8] = {
  { 1,-1,-1,-1,-1,-1,-1,-1},
  { 1,-1,-1, 1,-1, 1, 1,-1},
  { 1, 1, 1,-1,-1,-1, 1, 1},
  { 1,-1, 1, 1,-1,-1,-1, 1},
  { 1,-1, 1, 1, 1,-1,-1,-1},
  { 1,-1, 1,-1, 1, 1, 1,-1},
  { 1,-1,-1, 1,-1,-1, 1, 1},
  { 1, 1,-1,-1, 1, 1,-1, 1}};

// Build hamilton in WMMA B-fragment order, f16:
//   hamB[(((kt*2+hi)*8+tn)*16+n)*16+e] = hamilton[kt*32+16*hi+e][tn*16+n]
__global__ __launch_bounds__(256) void build_hamB(const float* __restrict__ weight,
                                                  _Float16* __restrict__ hamB) {
  int tid = blockIdx.x * blockDim.x + threadIdx.x;   // 65536 threads
  int e  = tid & 15;
  int n  = (tid >> 4) & 15;
  int tn = (tid >> 8) & 7;
  int hi = (tid >> 11) & 1;
  int kt = tid >> 12;
  int k = kt * 32 + 16 * hi + e;                     // hamilton row
  int i = k >> 6, r = k & 63;
  float v = OCT_SGN[tn][i] * weight[r * OUT_F + OCT_IDX[tn][i] * 16 + n];
  hamB[tid] = (_Float16)v;
}

// support = input(50000x512, f32) @ hamilton(512x128) via v_wmma_f32_16x16x32_f16.
// B is double-staged into 64KB LDS with CDNA5 async global->LDS copies so each
// WMMA depends on a short ds_load instead of an L2 round trip; A is streamed
// once, non-temporally.
__global__ __launch_bounds__(256) void gemm_wmma(const float* __restrict__ input,
                                                 const _Float16* __restrict__ hamB,
                                                 float* __restrict__ support) {
  __shared__ _Float16 shB[32768];                    // 64KB: kt-half of hamB
  const int lane = threadIdx.x & 31;
  const int wave = threadIdx.x >> 5;
  const int tm   = blockIdx.x * 8 + wave;            // 3125 row tiles (50000/16)
  const bool live = (tm < N_NODES / 16);             // wave-uniform
  const int hi = lane >> 4;
  const int mn = lane & 15;

  v8f acc[8] = {};

  const float* arow = input + (size_t)((live ? tm : 0) * 16 + mn) * IN_F;
  // LDS byte address of shB: low 32 bits of the flat shared-aperture pointer (ISA 10.2)
  const unsigned shbase = (unsigned)(size_t)(void*)shB;

  for (int p = 0; p < 2; ++p) {
    // ---- async copy 64KB of hamB into LDS: 256 threads x 16 x b128 (per-lane 16B) ----
    {
      const unsigned ldso = shbase + threadIdx.x * 16u;
      const char* g = (const char*)hamB + (size_t)p * 65536 + threadIdx.x * 16;
#pragma unroll
      for (int i = 0; i < 16; ++i) {
        asm volatile("global_load_async_to_lds_b128 %0, %1, off"
                     :: "v"(ldso + (unsigned)(i * 4096)), "v"(g + i * 4096) : "memory");
      }
      asm volatile("s_wait_asynccnt 0x0" ::: "memory");
    }
    __syncthreads();

    if (live) {                                      // wave-uniform: EXEC all-1 inside
#pragma unroll 2
      for (int kt8 = 0; kt8 < 8; ++kt8) {
        const int kb = (p * 8 + kt8) * 32;
        // A fragment: lane (m=mn, hi) holds K = [16hi,16hi+16)
        const v4f f0 = __builtin_nontemporal_load((const v4f*)(arow + kb + 16 * hi));
        const v4f f1 = __builtin_nontemporal_load((const v4f*)(arow + kb + 16 * hi + 4));
        const v4f f2 = __builtin_nontemporal_load((const v4f*)(arow + kb + 16 * hi + 8));
        const v4f f3 = __builtin_nontemporal_load((const v4f*)(arow + kb + 16 * hi + 12));
        v16h a;
        a[0]  = (_Float16)f0.x; a[1]  = (_Float16)f0.y; a[2]  = (_Float16)f0.z; a[3]  = (_Float16)f0.w;
        a[4]  = (_Float16)f1.x; a[5]  = (_Float16)f1.y; a[6]  = (_Float16)f1.z; a[7]  = (_Float16)f1.w;
        a[8]  = (_Float16)f2.x; a[9]  = (_Float16)f2.y; a[10] = (_Float16)f2.z; a[11] = (_Float16)f2.w;
        a[12] = (_Float16)f3.x; a[13] = (_Float16)f3.y; a[14] = (_Float16)f3.z; a[15] = (_Float16)f3.w;
        const _Float16* bp = shB + ((kt8 * 2 + hi) * 8) * 256 + mn * 16;
#pragma unroll
        for (int tn = 0; tn < 8; ++tn) {
          const v16h b = *(const v16h*)(bp + tn * 256);   // ds_load, ~LDS latency
          acc[tn] = __builtin_amdgcn_wmma_f32_16x16x32_f16(
              false, a, false, b, (short)0, acc[tn], false, false);
        }
      }
    }
    __syncthreads();                                 // before next stage overwrites shB
  }

  if (live) {
    // D layout: VGPR r = row (r + 8*hi), col mn -> half-wave coalesced stores
#pragma unroll
    for (int tn = 0; tn < 8; ++tn)
#pragma unroll
      for (int r = 0; r < 8; ++r)
        support[(size_t)(tm * 16 + r + 8 * hi) * OUT_F + tn * 16 + mn] = acc[tn][r];
  }
}

// ---- CSR build (counting sort by destination row): 1M int atomics vs 64M fp atomics ----

__global__ __launch_bounds__(256) void count_kernel(const int* __restrict__ erows,
                                                    int* __restrict__ counts) {
  int e = blockIdx.x * 256 + threadIdx.x;
  if (e < N_EDGES) atomicAdd(&counts[erows[e]], 1);
}

// Single-block exclusive scan over 50000 counts -> offsets[0..50000].
// Each chunk is fully staged in LDS before any write, so in-place (counts==offsets) is safe.
__global__ __launch_bounds__(1024) void scan_kernel(const int* __restrict__ counts,
                                                    int* __restrict__ offsets) {
  __shared__ int sh[1024];
  __shared__ int base;
  if (threadIdx.x == 0) base = 0;
  __syncthreads();
  for (int c0 = 0; c0 < N_NODES; c0 += 1024) {
    int i = c0 + (int)threadIdx.x;
    int v = (i < N_NODES) ? counts[i] : 0;
    sh[threadIdx.x] = v;
    __syncthreads();
    for (int d = 1; d < 1024; d <<= 1) {          // Hillis-Steele inclusive scan
      int t = (threadIdx.x >= (unsigned)d) ? sh[threadIdx.x - d] : 0;
      __syncthreads();
      sh[threadIdx.x] += t;
      __syncthreads();
    }
    int incl = sh[threadIdx.x];
    if (i < N_NODES) offsets[i] = base + incl - v;  // exclusive
    __syncthreads();
    if (threadIdx.x == 1023) base += sh[1023];
    __syncthreads();
  }
  if (threadIdx.x == 0) offsets[N_NODES] = base;    // == N_EDGES
}

__global__ __launch_bounds__(256) void scatter_kernel(const int* __restrict__ erows,
                                                      int* __restrict__ cursor,
                                                      int* __restrict__ edge_perm) {
  int e = blockIdx.x * 256 + threadIdx.x;
  if (e < N_EDGES) {
    int pos = atomicAdd(&cursor[erows[e]], 1);
    edge_perm[pos] = e;
  }
}

// Gather SpMM: one wave per row, float4 per lane, no output atomics.
__global__ __launch_bounds__(256) void gather_spmm(const float* __restrict__ support,
                                                   const float* __restrict__ evals,
                                                   const int* __restrict__ ecols,
                                                   const int* __restrict__ offsets,
                                                   const int* __restrict__ edge_perm,
                                                   float* __restrict__ out) {
  const int row  = (blockIdx.x * 256 + threadIdx.x) >> 5;   // 6250 blocks * 8 waves = 50000
  const int lane = threadIdx.x & 31;
  const int start = offsets[row];
  const int end   = offsets[row + 1];
  v4f acc = {0.f, 0.f, 0.f, 0.f};
  for (int i = start; i < end; ++i) {
    const int   e   = edge_perm[i];
    const float val = evals[e];
    const int   col = ecols[e];
    const v4f v = *(const v4f*)(support + (size_t)col * OUT_F + lane * 4);
    acc.x += val * v.x; acc.y += val * v.y; acc.z += val * v.z; acc.w += val * v.w;
  }
  *(v4f*)(out + (size_t)row * OUT_F + lane * 4) = acc;
}

// Per-column sum / sum-of-squares (coalesced; 128 threads = 128 columns per block)
__global__ __launch_bounds__(128) void stats_kernel(const float* __restrict__ out,
                                                    float* __restrict__ stats) {
  const int col = threadIdx.x;
  const int r0  = blockIdx.x * 250;
  const int r1  = (r0 + 250 < N_NODES) ? r0 + 250 : N_NODES;
  float s = 0.f, ss = 0.f;
  for (int r = r0; r < r1; ++r) {
    float x = out[(size_t)r * OUT_F + col];
    s  += x;
    ss += x * x;
  }
  unsafeAtomicAdd(&stats[col], s);
  unsafeAtomicAdd(&stats[OUT_F + col], ss);
}

// In-place batchnorm (biased var) + tanh
__global__ __launch_bounds__(256) void bn_tanh_kernel(float* __restrict__ out,
                                                      const float* __restrict__ stats,
                                                      const float* __restrict__ gamma,
                                                      const float* __restrict__ beta) {
  const int i = blockIdx.x * 256 + threadIdx.x;      // 6,400,000 = 25000*256 exactly
  const int col = i & (OUT_F - 1);
  const float invN = 1.0f / (float)N_NODES;
  const float mean = stats[col] * invN;
  const float var  = stats[OUT_F + col] * invN - mean * mean;
  const float x    = out[i];
  const float normed = (x - mean) * rsqrtf(var + BN_EPS) * gamma[col] + beta[col];
  out[i] = tanhf(normed);
}

extern "C" void kernel_launch(void* const* d_in, const int* in_sizes, int n_in,
                              void* d_out, int out_size, void* d_ws, size_t ws_size,
                              hipStream_t stream) {
  const float* input  = (const float*)d_in[0];
  const float* weight = (const float*)d_in[1];
  const float* gamma  = (const float*)d_in[2];
  const float* beta   = (const float*)d_in[3];
  const float* evals  = (const float*)d_in[4];
  const int*   erows  = (const int*)d_in[5];
  const int*   ecols  = (const int*)d_in[6];
  float* out = (float*)d_out;

  char* ws = (char*)d_ws;
  float*    support   = (float*)ws;                              // 25,600,000 B
  _Float16* hamB      = (_Float16*)(ws + 25600000);              //    131,072 B
  float*    stats     = (float*)   (ws + 25731072);              //      1,024 B
  int*      offsets   = (int*)     (ws + 25732096);              //    200,016 B (50001 ints)
  int*      cursor    = (int*)     (ws + 25932112);              //    200,000 B
  int*      edge_perm = (int*)     (ws + 26132112);              //  2,000,000 B  (total ~28.1 MB)

  // zero accumulation targets every call (graph-capture safe)
  hipMemsetAsync(stats,   0, 256 * sizeof(float), stream);
  hipMemsetAsync(offsets, 0, (N_NODES + 1) * sizeof(int), stream);

  build_hamB   <<<256,   256,  0, stream>>>(weight, hamB);
  gemm_wmma    <<<391,   256,  0, stream>>>(input, hamB, support);  // 391*8 waves >= 3125 tiles
  count_kernel <<<1954,  256,  0, stream>>>(erows, offsets);        // reuse offsets as counts
  scan_kernel  <<<1,     1024, 0, stream>>>(offsets, offsets);      // in-place safe (see kernel)
  hipMemcpyAsync(cursor, offsets, N_NODES * sizeof(int), hipMemcpyDeviceToDevice, stream);
  scatter_kernel<<<1954, 256,  0, stream>>>(erows, cursor, edge_perm);
  gather_spmm  <<<6250,  256,  0, stream>>>(support, evals, ecols, offsets, edge_perm, out);
  stats_kernel <<<200,   128,  0, stream>>>(out, stats);
  bn_tanh_kernel<<<25000,256,  0, stream>>>(out, stats, gamma, beta);
}
